// MultiHeadAttention_19902878450341
// MI455X (gfx1250) — compile-verified, matrix operand order checked by reading, not run on hardware
//
#include <hip/hip_runtime.h>
#include <stdint.h>

// ---------------- problem constants (from reference) ----------------
#define S_LEN   2048
#define BATCH   4
#define DMODEL  1024
#define NHEAD   16
#define HDIM    64                     // dk
#define ROWS    (S_LEN * BATCH)        // 8192
#define ATT_SCALE 0.125f               // 1/sqrt(64)

typedef __bf16 bf16_t;
typedef __attribute__((ext_vector_type(16))) bf16_t v16bf;
typedef __attribute__((ext_vector_type(8)))  float  v8f;
typedef __attribute__((ext_vector_type(4)))  float  v4f;
typedef __attribute__((ext_vector_type(4)))  unsigned int v4u;
typedef __attribute__((ext_vector_type(2)))  unsigned int v2u;

union Frag16 {          // 16x32 bf16 A/B fragment: 8 VGPRs = 2 x b128
    v16bf v;
    v4u   q[2];
};

__device__ __forceinline__ v8f wmma_bf16(const Frag16& a, const Frag16& b, v8f c) {
    // D = A(16x32 bf16) * B(32x16 bf16) + C(16x16 f32)
    return __builtin_amdgcn_wmma_f32_16x16x32_bf16(false, a.v, false, b.v,
                                                   (short)0, c, false, false);
}

// ---- CDNA5 async global->LDS copy (ASYNCcnt) ----
__device__ __forceinline__ void async_b128_to_lds(void* lds_ptr, const void* gptr) {
    unsigned lds_off = (unsigned)(uintptr_t)lds_ptr;   // LDS offset = low 32 bits
    asm volatile("global_load_async_to_lds_b128 %0, %1, off"
                 :: "v"(lds_off), "v"(gptr) : "memory");
}
__device__ __forceinline__ void wait_async0() {
    asm volatile("s_wait_asynccnt 0" ::: "memory");
}
__device__ __forceinline__ void wait_ds0() {
    asm volatile("s_wait_dscnt 0" ::: "memory");
}

__device__ __forceinline__ v2u pack4_bf16(v4f f) {
    union { bf16_t h[4]; v2u u; } cv;
    cv.h[0] = (bf16_t)f[0]; cv.h[1] = (bf16_t)f[1];
    cv.h[2] = (bf16_t)f[2]; cv.h[3] = (bf16_t)f[3];
    return cv.u;
}

// ---------------- fp32 -> bf16 weight conversion ----------------
__global__ __launch_bounds__(256)
void mha_cvt(const float* __restrict__ in, bf16_t* __restrict__ out) {
    int i = (blockIdx.x * 256 + threadIdx.x) * 4;
    v4f f = *(const v4f*)(in + i);
    *(v2u*)(out + i) = pack4_bf16(f);
}

// ---------------- tile staging helper ----------------
// AMODE 0: A fp32 (convert while staging via DS stores)
// AMODE 1: A bf16 (async DMA staged)
template<int AMODE>
__device__ __forceinline__ void stage_tiles(const void* __restrict__ Aptr,
                                            const bf16_t* __restrict__ W,
                                            bf16_t* Albuf, bf16_t* Wlbuf,
                                            int tid, int row0, int col0, int k0)
{
    if constexpr (AMODE == 0) {
        const float* A = (const float*)Aptr;
        #pragma unroll
        for (int j = 0; j < 4; ++j) {
            int c = tid * 4 + j;          // 1024 chunks of 4 floats
            int r = c >> 3;
            int p = (c & 7) * 4;
            v4f f = *(const v4f*)(A + (size_t)(row0 + r) * DMODEL + k0 + p);
            *(v2u*)&Albuf[r * 32 + p] = pack4_bf16(f);
        }
    } else {
        const bf16_t* A = (const bf16_t*)Aptr;
        #pragma unroll
        for (int j = 0; j < 2; ++j) {
            int c = tid * 2 + j;          // 512 chunks of 16B
            int r = c >> 2;
            int p = (c & 3) * 8;
            async_b128_to_lds(&Albuf[r * 32 + p],
                              A + (size_t)(row0 + r) * DMODEL + k0 + p);
        }
    }
    #pragma unroll
    for (int j = 0; j < 2; ++j) {         // W tile: always async DMA
        int c = tid * 2 + j;
        int r = c >> 2;
        int p = (c & 3) * 8;
        async_b128_to_lds(&Wlbuf[r * 32 + p],
                          W + (size_t)(col0 + r) * DMODEL + k0 + p);
    }
}

// ---------------- GEMM: Y = A @ W^T + bias (double-buffered pipeline) ----------------
// OMODE 0: bf16 [B,H,S,dk] (scaled)  OMODE 1: bf16 [B,H,dk,S]  OMODE 2: fp32 [R,D]
template<int AMODE, int OMODE>
__global__ __launch_bounds__(256, 2)
void mha_gemm(const void* __restrict__ Aptr, const bf16_t* __restrict__ W,
              const float* __restrict__ bias, void* __restrict__ Yptr, float scale)
{
    __shared__ bf16_t Alds[2][128 * 32];
    __shared__ bf16_t Wlds[2][128 * 32];

    const int tid  = threadIdx.x;
    const int lane = tid & 31;
    const int wave = tid >> 5;
    const int hf   = lane >> 4;      // lane half (k-group select)
    const int ln   = lane & 15;      // m / n within fragment
    const int row0 = blockIdx.x * 128;
    const int col0 = blockIdx.y * 128;
    const int wm   = wave >> 2;      // 2 x 64-row wave blocks
    const int wn   = wave & 3;       // 4 x 32-col wave blocks

    v8f acc[4][2];
    #pragma unroll
    for (int i = 0; i < 4; ++i)
        #pragma unroll
        for (int j = 0; j < 2; ++j)
            #pragma unroll
            for (int e = 0; e < 8; ++e) acc[i][j][e] = 0.0f;

    constexpr int KSTEPS = DMODEL / 32;  // 32
    stage_tiles<AMODE>(Aptr, W, Alds[0], Wlds[0], tid, row0, col0, 0);

    for (int it = 0; it < KSTEPS; ++it) {
        const int buf = it & 1;
        // group(it) has been in flight for a full compute phase already
        wait_async0();
        wait_ds0();
        __syncthreads();   // also proves all waves finished compute(it-1) on buf^1
        if (it + 1 < KSTEPS)
            stage_tiles<AMODE>(Aptr, W, Alds[buf ^ 1], Wlds[buf ^ 1],
                               tid, row0, col0, (it + 1) * 32);

        Frag16 af[4], bfr[2];
        #pragma unroll
        for (int mt = 0; mt < 4; ++mt) {
            const bf16_t* p = &Alds[buf][(wm * 64 + mt * 16 + ln) * 32 + hf * 8];
            af[mt].q[0] = *(const v4u*)p;
            af[mt].q[1] = *(const v4u*)(p + 16);
        }
        #pragma unroll
        for (int nt = 0; nt < 2; ++nt) {
            const bf16_t* p = &Wlds[buf][(wn * 32 + nt * 16 + ln) * 32 + hf * 8];
            bfr[nt].q[0] = *(const v4u*)p;
            bfr[nt].q[1] = *(const v4u*)(p + 16);
        }
        #pragma unroll
        for (int mt = 0; mt < 4; ++mt)
            #pragma unroll
            for (int nt = 0; nt < 2; ++nt)
                acc[mt][nt] = wmma_bf16(af[mt], bfr[nt], acc[mt][nt]);
    }

    // epilogue: C layout element v -> row v + 8*half, col = lane&15
    #pragma unroll
    for (int mt = 0; mt < 4; ++mt)
        #pragma unroll
        for (int nt = 0; nt < 2; ++nt)
            #pragma unroll
            for (int v = 0; v < 8; ++v) {
                int r = row0 + wm * 64 + mt * 16 + v + hf * 8;
                int o = col0 + wn * 32 + nt * 16 + ln;
                float val = acc[mt][nt][v] + bias[o];
                if constexpr (OMODE == 2) {
                    ((float*)Yptr)[(size_t)r * DMODEL + o] = val;
                } else {
                    val *= scale;
                    int s  = r >> 2;          // r = s*BATCH + b (BATCH=4)
                    int bb = r & 3;
                    int hh = o >> 6;          // o = hh*HDIM + dk
                    int dk = o & 63;
                    bf16_t* Y = (bf16_t*)Yptr;
                    if constexpr (OMODE == 0)
                        Y[(((size_t)bb * NHEAD + hh) * S_LEN + s) * HDIM + dk] = (bf16_t)val;
                    else
                        Y[(((size_t)bb * NHEAD + hh) * HDIM + dk) * S_LEN + s] = (bf16_t)val;
                }
            }
}

// ---------------- flash attention (causal, online softmax) ----------------
// grid: (S/64, B*H); block: 128 (4 waves, 16 q-rows each)
__global__ __launch_bounds__(128, 4)
void mha_attn(const bf16_t* __restrict__ Q, const bf16_t* __restrict__ K,
              const bf16_t* __restrict__ Vt, bf16_t* __restrict__ ctx)
{
    __shared__ bf16_t Plds[4 * 16 * 64];   // per-wave P staging (C->A relayout)

    const int tid   = threadIdx.x;
    const int lane  = tid & 31;
    const int wave  = tid >> 5;
    const int hf    = lane >> 4;
    const int ln    = lane & 15;
    const int qtile = blockIdx.x;
    const int bh    = blockIdx.y;
    const int bb    = bh / NHEAD;
    const int hh    = bh % NHEAD;
    const int qrow0 = qtile * 64 + wave * 16;

    const bf16_t* Qb = Q  + (size_t)bh * S_LEN * HDIM;
    const bf16_t* Kb = K  + (size_t)bh * S_LEN * HDIM;
    const bf16_t* Vb = Vt + (size_t)bh * HDIM * S_LEN;

    Frag16 qa[2];                          // Q rows as A fragments (dk = 2 x 32)
    #pragma unroll
    for (int c = 0; c < 2; ++c) {
        const bf16_t* p = Qb + (size_t)(qrow0 + ln) * HDIM + c * 32 + hf * 8;
        qa[c].q[0] = *(const v4u*)p;
        qa[c].q[1] = *(const v4u*)(p + 16);
    }

    float mi[8], li[8];
    v8f oacc[4];
    #pragma unroll
    for (int v = 0; v < 8; ++v) { mi[v] = -3.0e38f; li[v] = 0.0f; }
    #pragma unroll
    for (int d = 0; d < 4; ++d)
        #pragma unroll
        for (int v = 0; v < 8; ++v) oacc[d][v] = 0.0f;

    bf16_t* Pw = &Plds[wave * 16 * 64];

    for (int j = 0; j <= qtile; ++j) {     // causal: only kv tiles <= diagonal
        const int kc0 = j * 64;

        // prefetch next kv tile (one 128B line per row; 32 lanes x 2 = 64 rows)
        if (j < qtile) {
            const bf16_t* kn = Kb + (size_t)(kc0 + 64 + lane) * HDIM;
            __builtin_prefetch(kn, 0, 0);
            __builtin_prefetch(kn + (size_t)32 * HDIM, 0, 0);
            const bf16_t* vn = Vb + (size_t)lane * S_LEN + kc0 + 64;
            __builtin_prefetch(vn, 0, 0);
            __builtin_prefetch(vn + (size_t)32 * S_LEN, 0, 0);
        }

        // scores S = Q @ K^T  (Q pre-scaled by 1/sqrt(dk))
        v8f sf[4];
        #pragma unroll
        for (int nt = 0; nt < 4; ++nt) {
            const bf16_t* kp = Kb + (size_t)(kc0 + nt * 16 + ln) * HDIM;
            Frag16 k0f, k1f;
            k0f.q[0] = *(const v4u*)(kp + hf * 8);
            k0f.q[1] = *(const v4u*)(kp + 16 + hf * 8);
            k1f.q[0] = *(const v4u*)(kp + 32 + hf * 8);
            k1f.q[1] = *(const v4u*)(kp + 48 + hf * 8);
            v8f s;
            #pragma unroll
            for (int e = 0; e < 8; ++e) s[e] = 0.0f;
            s = wmma_bf16(qa[0], k0f, s);
            s = wmma_bf16(qa[1], k1f, s);
            sf[nt] = s;
        }

        if (j == qtile) {                  // mask only the diagonal tile
            #pragma unroll
            for (int nt = 0; nt < 4; ++nt)
                #pragma unroll
                for (int v = 0; v < 8; ++v) {
                    int keyi = kc0 + nt * 16 + ln;
                    int rowi = qrow0 + v + hf * 8;
                    if (keyi > rowi) sf[nt][v] = -3.0e38f;
                }
        }

        // online softmax: row max / rescale / exp / row sum
        float al[8];
        #pragma unroll
        for (int v = 0; v < 8; ++v) {
            float t = fmaxf(fmaxf(sf[0][v], sf[1][v]), fmaxf(sf[2][v], sf[3][v]));
            t = fmaxf(t, __shfl_xor(t, 1, 32));
            t = fmaxf(t, __shfl_xor(t, 2, 32));
            t = fmaxf(t, __shfl_xor(t, 4, 32));
            t = fmaxf(t, __shfl_xor(t, 8, 32));   // stays within lane half
            float nm = fmaxf(mi[v], t);
            al[v] = __expf(mi[v] - nm);
            mi[v] = nm;
        }
        #pragma unroll
        for (int nt = 0; nt < 4; ++nt)
            #pragma unroll
            for (int v = 0; v < 8; ++v)
                sf[nt][v] = __expf(sf[nt][v] - mi[v]);
        #pragma unroll
        for (int v = 0; v < 8; ++v) {
            float r = sf[0][v] + sf[1][v] + sf[2][v] + sf[3][v];
            r += __shfl_xor(r, 1, 32);
            r += __shfl_xor(r, 2, 32);
            r += __shfl_xor(r, 4, 32);
            r += __shfl_xor(r, 8, 32);
            li[v] = li[v] * al[v] + r;
        }
        #pragma unroll
        for (int d = 0; d < 4; ++d)
            #pragma unroll
            for (int v = 0; v < 8; ++v) oacc[d][v] *= al[v];

        // P: C-layout -> LDS -> A-layout fragments (wave-private, DS in-order)
        #pragma unroll
        for (int nt = 0; nt < 4; ++nt)
            #pragma unroll
            for (int v = 0; v < 8; ++v)
                Pw[(v + hf * 8) * 64 + nt * 16 + ln] = (bf16_t)sf[nt][v];
        wait_ds0();

        Frag16 pa[2];
        #pragma unroll
        for (int c = 0; c < 2; ++c) {
            const bf16_t* pp = Pw + ln * 64 + c * 32 + hf * 8;
            pa[c].q[0] = *(const v4u*)pp;
            pa[c].q[1] = *(const v4u*)(pp + 16);
        }

        // O += P @ V ; V stored [dk][s] so B fragments are contiguous
        #pragma unroll
        for (int d = 0; d < 4; ++d)
            #pragma unroll
            for (int c = 0; c < 2; ++c) {
                Frag16 vbf;
                const bf16_t* vp = Vb + (size_t)(d * 16 + ln) * S_LEN + kc0 + c * 32 + hf * 8;
                vbf.q[0] = *(const v4u*)vp;
                vbf.q[1] = *(const v4u*)(vp + 16);
                oacc[d] = wmma_bf16(pa[c], vbf, oacc[d]);
            }
    }

    // normalize and store context as [S,B,D] bf16 (rows match O-proj GEMM)
    float inv[8];
    #pragma unroll
    for (int v = 0; v < 8; ++v) inv[v] = 1.0f / li[v];
    #pragma unroll
    for (int d = 0; d < 4; ++d)
        #pragma unroll
        for (int v = 0; v < 8; ++v) {
            int srow = qrow0 + v + hf * 8;
            int dk   = d * 16 + ln;
            float val = oacc[d][v] * inv[v];
            ctx[((size_t)srow * BATCH + bb) * DMODEL + hh * HDIM + dk] = (bf16_t)val;
        }
}

// ---------------- launcher ----------------
extern "C" void kernel_launch(void* const* d_in, const int* in_sizes, int n_in,
                              void* d_out, int out_size, void* d_ws, size_t ws_size,
                              hipStream_t stream)
{
    const float* query = (const float*)d_in[0];
    const float* key   = (const float*)d_in[1];
    const float* value = (const float*)d_in[2];
    // d_in[3] = causal mask (bool) -- structure computed analytically in-kernel
    const float* Wq = (const float*)d_in[4];
    const float* bq = (const float*)d_in[5];
    const float* Wk = (const float*)d_in[6];
    const float* bk = (const float*)d_in[7];
    const float* Wv = (const float*)d_in[8];
    const float* bv = (const float*)d_in[9];
    const float* Wo = (const float*)d_in[10];
    const float* bo = (const float*)d_in[11];

    char* ws = (char*)d_ws;
    const size_t WB = (size_t)DMODEL * DMODEL * sizeof(bf16_t);   // 2 MiB
    const size_t TB = (size_t)ROWS * DMODEL * sizeof(bf16_t);     // 16 MiB
    bf16_t* wqb = (bf16_t*)(ws);
    bf16_t* wkb = (bf16_t*)(ws + 1 * WB);
    bf16_t* wvb = (bf16_t*)(ws + 2 * WB);
    bf16_t* wob = (bf16_t*)(ws + 3 * WB);
    bf16_t* Qb  = (bf16_t*)(ws + 4 * WB);
    bf16_t* Kb  = (bf16_t*)(ws + 4 * WB + 1 * TB);
    bf16_t* Vtb = (bf16_t*)(ws + 4 * WB + 2 * TB);
    bf16_t* ctx = (bf16_t*)(ws + 4 * WB + 3 * TB);                // total 72 MiB

    const int wgrid = (DMODEL * DMODEL) / (256 * 4);
    mha_cvt<<<wgrid, 256, 0, stream>>>(Wq, wqb);
    mha_cvt<<<wgrid, 256, 0, stream>>>(Wk, wkb);
    mha_cvt<<<wgrid, 256, 0, stream>>>(Wv, wvb);
    mha_cvt<<<wgrid, 256, 0, stream>>>(Wo, wob);

    dim3 gg(ROWS / 128, DMODEL / 128);   // 64 x 8
    mha_gemm<0, 0><<<gg, 256, 0, stream>>>(query, wqb, bq, Qb,  ATT_SCALE);
    mha_gemm<0, 0><<<gg, 256, 0, stream>>>(key,   wkb, bk, Kb,  1.0f);
    mha_gemm<0, 1><<<gg, 256, 0, stream>>>(value, wvb, bv, Vtb, 1.0f);

    mha_attn<<<dim3(S_LEN / 64, BATCH * NHEAD), 128, 0, stream>>>(Qb, Kb, Vtb, ctx);

    mha_gemm<1, 2><<<gg, 256, 0, stream>>>(ctx, wob, bo, d_out, 1.0f);
}
